// TopKPrunedAttention_19095424598883
// MI455X (gfx1250) — matrix operand order repaired and checked by reading, not compiled, not run on hardware
//
#include <hip/hip_runtime.h>
#include <hip/hip_bf16.h>
#include <math.h>

typedef __attribute__((ext_vector_type(16))) _Float16 v16h;
typedef __attribute__((ext_vector_type(8)))  float    v8f;

#define Bn   8
#define Sn   8192
#define Dn   4096
#define HQ   32
#define HKV  8
#define DH   128
#define KK   128     // K_KEEP
#define KSPLIT 8     // k-dimension split for the skinny GEMMs
#define KCHUNK (Dn / KSPLIT)   // 512

// ---------------------------------------------------------------------------
// 1) importance[b,s] = mean over 32 heads of attn_w[b,h,0,s]
// ---------------------------------------------------------------------------
__global__ __launch_bounds__(256) void importance_kernel(
    const float* __restrict__ attn, float* __restrict__ imp) {
  int idx = blockIdx.x * 256 + threadIdx.x;        // 0 .. B*S-1 = 65535
  int b = idx >> 13;
  int s = idx & (Sn - 1);
  const float* p = attn + ((size_t)b * HQ) * Sn + s;
  float acc = 0.f;
#pragma unroll
  for (int h = 0; h < HQ; ++h) acc += p[(size_t)h * Sn];
  imp[idx] = acc * (1.f / 32.f);
}

// ---------------------------------------------------------------------------
// 2) top-128 per batch, descending (ties -> lowest index, matching lax.top_k)
// ---------------------------------------------------------------------------
__global__ __launch_bounds__(256) void topk_kernel(
    const float* __restrict__ imp, int* __restrict__ topk) {
  __shared__ float vals[Sn];
  __shared__ float rv[256];
  __shared__ int   ri[256];
  const int b = blockIdx.x;
  const int t = threadIdx.x;
  for (int i = t; i < Sn; i += 256) vals[i] = imp[(size_t)b * Sn + i];
  __syncthreads();
  for (int sel = 0; sel < KK; ++sel) {
    float bv = -INFINITY; int bi = 0x7fffffff;
    for (int i = t; i < Sn; i += 256) {
      float v = vals[i];
      if (v > bv || (v == bv && i < bi)) { bv = v; bi = i; }
    }
    rv[t] = bv; ri[t] = bi;
    __syncthreads();
    for (int off = 128; off > 0; off >>= 1) {
      if (t < off) {
        float v2 = rv[t + off]; int i2 = ri[t + off];
        if (v2 > rv[t] || (v2 == rv[t] && i2 < ri[t])) { rv[t] = v2; ri[t] = i2; }
      }
      __syncthreads();
    }
    if (t == 0) { topk[b * KK + sel] = ri[0]; vals[ri[0]] = -INFINITY; }
    __syncthreads();
  }
}

// ---------------------------------------------------------------------------
// 3) skinny GEMM via WMMA: Out[m,n] = sum_k A[m,k] * Wm[n,k]
//    M=8 padded to 16 by DUPLICATING rows 0..7 into 8..15 (rows never mix in
//    D = A x B, so duplicates are harmless and we avoid zero-fill divergence).
//    grid.x: 32 blocks x 8 waves -> 256 column tiles of 16
//    grid.y: KSPLIT partial sums -> partials[s][m][n], reduced afterwards
// ---------------------------------------------------------------------------

// pack 16 contiguous f32 -> v16h  (unused; the B operand is packed with pack_a16)
__device__ __forceinline__ v16h pack_b16(const float* __restrict__ p) {
  const float4* fp = reinterpret_cast<const float4*>(p);
  float4 x0 = fp[0], x1 = fp[1], x2 = fp[2], x3 = fp[3];
  v16h h;
  h[0]  = (_Float16)x0.x; h[1]  = (_Float16)x0.y;
  h[2]  = (_Float16)x0.z; h[3]  = (_Float16)x0.w;
  h[4]  = (_Float16)x1.x; h[5]  = (_Float16)x1.y;
  h[6]  = (_Float16)x1.z; h[7]  = (_Float16)x1.w;
  h[8]  = (_Float16)x2.x; h[9]  = (_Float16)x2.y;
  h[10] = (_Float16)x2.z; h[11] = (_Float16)x2.w;
  h[12] = (_Float16)x3.x; h[13] = (_Float16)x3.y;
  h[14] = (_Float16)x3.z; h[15] = (_Float16)x3.w;
  return h;
}

// pack A operand: elems 0..7 = p[0..7], elems 8..15 = p[16..23]
// (A layout: i<8 -> K=k0+8g+i ; i>=8 -> K=k0+16+8g+(i-8);  p = arow+k0+8g)
__device__ __forceinline__ v16h pack_a16(const float* __restrict__ p) {
  const float4* fp  = reinterpret_cast<const float4*>(p);
  const float4* fp2 = reinterpret_cast<const float4*>(p + 16);
  float4 x0 = fp[0],  x1 = fp[1];
  float4 x2 = fp2[0], x3 = fp2[1];
  v16h h;
  h[0]  = (_Float16)x0.x; h[1]  = (_Float16)x0.y;
  h[2]  = (_Float16)x0.z; h[3]  = (_Float16)x0.w;
  h[4]  = (_Float16)x1.x; h[5]  = (_Float16)x1.y;
  h[6]  = (_Float16)x1.z; h[7]  = (_Float16)x1.w;
  h[8]  = (_Float16)x2.x; h[9]  = (_Float16)x2.y;
  h[10] = (_Float16)x2.z; h[11] = (_Float16)x2.w;
  h[12] = (_Float16)x3.x; h[13] = (_Float16)x3.y;
  h[14] = (_Float16)x3.z; h[15] = (_Float16)x3.w;
  return h;
}

__global__ __launch_bounds__(256) void gemm_a8_bt_kernel(
    const float* __restrict__ A,     // [8, 4096]
    const float* __restrict__ Wm,    // [4096, 4096] row-major, used transposed
    float* __restrict__ partials) {  // [KSPLIT][8][4096]
  const int lane = threadIdx.x & 31;
  const int wid  = threadIdx.x >> 5;
  const int gw   = blockIdx.x * 8 + wid;   // 0..255
  const int n0   = gw * 16;
  const int m    = lane & 15;              // B column / logical A row
  const int g    = lane >> 4;              // lane half
  const int kbeg = blockIdx.y * KCHUNK;

  const float* brow = Wm + (size_t)(n0 + m) * Dn;
  const float* arow = A + (size_t)(m & 7) * Dn;   // duplicate rows 0..7 -> 8..15

  v8f c0 = {0.f, 0.f, 0.f, 0.f, 0.f, 0.f, 0.f, 0.f};
  v8f c1 = {0.f, 0.f, 0.f, 0.f, 0.f, 0.f, 0.f, 0.f};

  // two independent accumulator chains -> interleaved load/cvt/wmma pipelines
  for (int k0 = kbeg; k0 < kbeg + KCHUNK; k0 += 64) {
    __builtin_prefetch((const void*)(brow + k0 + 256), 0, 1);
    v16h bh0 = pack_a16(brow + k0 + g * 8);
    v16h ah0 = pack_a16(arow + k0 + g * 8);
    v16h bh1 = pack_a16(brow + k0 + 32 + g * 8);
    v16h ah1 = pack_a16(arow + k0 + 32 + g * 8);
    c0 = __builtin_amdgcn_wmma_f32_16x16x32_f16(
        false, ah0, false, bh0, (short)0, c0, false, false);
    c1 = __builtin_amdgcn_wmma_f32_16x16x32_f16(
        false, ah1, false, bh1, (short)0, c1, false, false);
  }

#pragma unroll
  for (int j = 0; j < 8; ++j) c0[j] += c1[j];

  if (g == 0) {   // C VGPR j holds row M=j for lanes 0..15; only M<8 is real
    float* dst = partials + (size_t)blockIdx.y * (8 * Dn) + n0 + m;
#pragma unroll
    for (int j = 0; j < 8; ++j) dst[(size_t)j * Dn] = c0[j];
  }
}

__global__ __launch_bounds__(256) void ksplit_reduce_kernel(
    const float* __restrict__ partials, float* __restrict__ dst) {
  int idx = blockIdx.x * 256 + threadIdx.x;   // 0 .. 8*4096-1
  float acc = 0.f;
#pragma unroll
  for (int s = 0; s < KSPLIT; ++s) acc += partials[(size_t)s * (8 * Dn) + idx];
  dst[idx] = acc;
}

// ---------------------------------------------------------------------------
// 4) pruned attention per (b, kv_head): gather K -> scores -> softmax -> ctx
//    128 threads; LDS rows padded to 129 floats (conflict-free on 64 banks)
// ---------------------------------------------------------------------------
__global__ __launch_bounds__(128) void attn_kernel(
    const float* __restrict__ kin, const float* __restrict__ vin,
    const float* __restrict__ q,   const int* __restrict__ topk,
    float* __restrict__ ctx, float* __restrict__ probs_out) {
  extern __shared__ float sm[];
  float* kv_sh = sm;                    // 128 * 129
  float* q_sh  = sm + 128 * 129;        // 4 * 128
  float* p_sh  = q_sh + 4 * 128;        // 4 * 128
  float* red   = p_sh + 4 * 128;        // 128
  int*   rows  = (int*)(red + 128);     // 128

  const int b  = blockIdx.x >> 3;
  const int hk = blockIdx.x & 7;
  const int t  = threadIdx.x;           // 0..127

  rows[t] = topk[b * KK + t];
#pragma unroll
  for (int r = 0; r < 4; ++r)
    q_sh[r * 128 + t] = q[(size_t)b * Dn + (hk * 4 + r) * DH + t];
  __syncthreads();

  const float* kbase = kin + ((size_t)b * HKV + hk) * Sn * DH;
  for (int j = 0; j < KK; ++j)
    kv_sh[j * 129 + t] = kbase[(size_t)rows[j] * DH + t];
  __syncthreads();

  const float scale = 0.08838834764831845f;   // 1/sqrt(128)
#pragma unroll 1
  for (int r = 0; r < 4; ++r) {
    float s = 0.f;
#pragma unroll 8
    for (int d = 0; d < DH; ++d) s += q_sh[r * 128 + d] * kv_sh[t * 129 + d];
    s *= scale;
    red[t] = s; __syncthreads();
    for (int off = 64; off > 0; off >>= 1) {
      if (t < off) red[t] = fmaxf(red[t], red[t + off]);
      __syncthreads();
    }
    float mx = red[0]; __syncthreads();
    float e = __expf(s - mx);
    red[t] = e; __syncthreads();
    for (int off = 64; off > 0; off >>= 1) {
      if (t < off) red[t] += red[t + off];
      __syncthreads();
    }
    float inv = 1.f / red[0]; __syncthreads();
    float p = e * inv;
    p_sh[r * 128 + t] = p;
    probs_out[((size_t)b * HQ + hk * 4 + r) * KK + t] = p;
  }
  __syncthreads();

  const float* vbase = vin + ((size_t)b * HKV + hk) * Sn * DH;
  for (int j = 0; j < KK; ++j)
    kv_sh[j * 129 + t] = vbase[(size_t)rows[j] * DH + t];
  __syncthreads();

#pragma unroll 1
  for (int r = 0; r < 4; ++r) {
    float acc = 0.f;
#pragma unroll 8
    for (int tok = 0; tok < KK; ++tok)
      acc += p_sh[r * 128 + tok] * kv_sh[tok * 129 + t];
    ctx[(size_t)b * Dn + (hk * 4 + r) * DH + t] = acc;
  }
}

// ---------------------------------------------------------------------------
// launch
// ---------------------------------------------------------------------------
extern "C" void kernel_launch(void* const* d_in, const int* in_sizes, int n_in,
                              void* d_out, int out_size, void* d_ws, size_t ws_size,
                              hipStream_t stream) {
  const float* hidden = (const float*)d_in[0];   // [8,1,4096]
  const float* attnw  = (const float*)d_in[1];   // [8,32,1,8192]
  const float* kin    = (const float*)d_in[2];   // [8,8,8192,128]
  const float* vin    = (const float*)d_in[3];   // [8,8,8192,128]
  const float* Wq     = (const float*)d_in[4];   // [4096,4096]
  const float* Wo     = (const float*)d_in[5];   // [4096,4096]

  float* out   = (float*)d_out;          // [8,1,4096] = 32768 floats
  float* probs = out + Bn * Dn;          // [8,32,1,128] = 32768 floats

  float* ws   = (float*)d_ws;
  float* imp  = ws;                       // 65536 floats
  float* qws  = ws + 65536;               // 32768
  float* ctx  = ws + 98304;               // 32768
  float* part = ws + 131072;              // KSPLIT * 32768 = 262144
  int*   topk = (int*)(ws + 393216);      // 1024 ints

  importance_kernel<<<(Bn * Sn) / 256, 256, 0, stream>>>(attnw, imp);
  topk_kernel<<<Bn, 256, 0, stream>>>(imp, topk);

  gemm_a8_bt_kernel<<<dim3(32, KSPLIT), 256, 0, stream>>>(hidden, Wq, part);
  ksplit_reduce_kernel<<<(Bn * Dn) / 256, 256, 0, stream>>>(part, qws);

  size_t smem = (size_t)(128 * 129 + 4 * 128 + 4 * 128 + 128) * sizeof(float)
              + 128 * sizeof(int);
  attn_kernel<<<Bn * HKV, 128, smem, stream>>>(kin, vin, qws, topk, ctx, probs);

  gemm_a8_bt_kernel<<<dim3(32, KSPLIT), 256, 0, stream>>>(ctx, Wo, part);
  ksplit_reduce_kernel<<<(Bn * Dn) / 256, 256, 0, stream>>>(part, out);
}
